// Discriminator_4509715661128
// MI455X (gfx1250) — matrix operand order repaired and checked by reading, not compiled, not run on hardware
//
#include <hip/hip_runtime.h>

// ---------------------------------------------------------------------------
// Edge-MLP discriminator on gfx1250 (MI455X).
// bf16 WMMA (f32 accum), weights = A-matrix (LDS-resident), features = B.
// 2 edge-tiles per wave iteration -> each weight fragment load feeds 2 WMMAs.
// Layer 3 is an in-register dot + cross-half shuffle (no LDS round-trip).
// ---------------------------------------------------------------------------

typedef __attribute__((ext_vector_type(16))) __bf16 v16bf;
typedef __attribute__((ext_vector_type(8)))  __bf16 v8bf;
typedef __attribute__((ext_vector_type(8)))  float  v8f;

// LDS layout (bytes)
#define LDS_W1    0        // 128 frags * 1024B (A layout, 16 nt * 8 kt)
#define LDS_W2    131072   // 32 frags * 1024B  (A layout, 4 nt * 8 kt)
#define LDS_B1    163840   // 256 f32
#define LDS_B2    164864   // 64 f32
#define LDS_H1    165120   // per-wave: 2 tiles * 16 edges * 272 bf16
#define H1_ROWB   544      // 272 bf16 per row (bank-conflict padding)
#define H1_TILEB  8704     // 16 rows
#define H1_WAVEB  17408    // 2 tiles
#define LDS_TOTAL 304384   // 297.25 KB < 320 KB/WGP

#define WS_ZB_OFF 163840   // byte offset of bf16 z copy inside d_ws

__device__ __forceinline__ __bf16 bf(float x) { return (__bf16)x; }

__device__ __forceinline__ v16bf cat8(v8bf lo, v8bf hi) {
  return __builtin_shufflevector(lo, hi, 0,1,2,3,4,5,6,7,8,9,10,11,12,13,14,15);
}

// ---------------------------------------------------------------------------
// Prep 1: swizzle W1 [256,256] / W2 [64,256] ([out,in] f32) into bf16 WMMA
// A-fragments. Fragment (nt,kt) = 1024B: chunk c in {0,1} at c*512 + lane*16.
// A lane layout: M(row)=lane&15, K = i + 16*half; k = kt*32 + K.
// ---------------------------------------------------------------------------
__global__ void prep_weights(const float* __restrict__ W1,
                             const float* __restrict__ W2,
                             __bf16* __restrict__ wsbf) {
  int q = blockIdx.x * blockDim.x + threadIdx.x;
  if (q >= 81920) return;                 // 65536 (W1) + 16384 (W2) bf16
  const float* W = (q < 65536) ? W1 : W2;
  int idx = (q < 65536) ? q : (q - 65536);
  int fragIdx = idx >> 9;                 // 512 bf16 per fragment
  int within  = idx & 511;
  int c    = within >> 8;
  int lane = (within >> 3) & 31;
  int j    = within & 7;
  int i    = c * 8 + j;
  int kt   = fragIdx & 7;
  int nt   = fragIdx >> 3;
  int half = lane >> 4;
  int nl   = lane & 15;
  int K = i + 16 * half;                       // A-matrix K mapping
  int k = kt * 32 + K;
  int n = nt * 16 + nl;
  wsbf[q] = bf(W[n * 256 + k]);
}

// ---------------------------------------------------------------------------
// Prep 2: z (f32, [N,128]) -> bf16 row-major copy in d_ws.
// ---------------------------------------------------------------------------
__global__ void prep_z(const float* __restrict__ z, __bf16* __restrict__ zb,
                       int n) {
  int idx = (blockIdx.x * blockDim.x + threadIdx.x) * 8;
  if (idx >= n) return;
  float4 a = *(const float4*)(z + idx);
  float4 b = *(const float4*)(z + idx + 4);
  v8bf o;
  o[0] = bf(a.x); o[1] = bf(a.y); o[2] = bf(a.z); o[3] = bf(a.w);
  o[4] = bf(b.x); o[5] = bf(b.y); o[6] = bf(b.z); o[7] = bf(b.w);
  *(v8bf*)(zb + idx) = o;
}

// ---------------------------------------------------------------------------
// Main: each wave owns TWO 16-edge tiles per iteration.
// ---------------------------------------------------------------------------
__global__ void __launch_bounds__(256)
edge_mlp_kernel(const __bf16* __restrict__ zb,      // bf16 z in d_ws
                const long long* __restrict__ ei,   // [2, E] int64
                const float* __restrict__ b1,
                const float* __restrict__ b2,
                const float* __restrict__ W3,
                const float* __restrict__ b3,
                const __bf16* __restrict__ wfrag,   // prepped fragments in d_ws
                float* __restrict__ out,
                int E, int nTiles) {
  extern __shared__ char smem[];

  // --- cooperative fill: 160KB weights + b1/b2 into LDS ---
  {
    const uint4* src = (const uint4*)wfrag;
    uint4* dst = (uint4*)smem;
    for (int t = threadIdx.x; t < 10240; t += 256) dst[t] = src[t];
    float* b1s = (float*)(smem + LDS_B1);
    b1s[threadIdx.x] = b1[threadIdx.x];          // 256 floats
    float* b2s = (float*)(smem + LDS_B2);
    if (threadIdx.x < 64) b2s[threadIdx.x] = b2[threadIdx.x];
  }
  __syncthreads();

  const int wave = threadIdx.x >> 5;
  const int lane = threadIdx.x & 31;
  const int nl   = lane & 15;
  const int half = lane >> 4;

  const float b3v = b3[0];

  // per-lane W3 values matching the D layout: feature nt*16 + 8*half + r
  float w3v[32];
  #pragma unroll
  for (int nt = 0; nt < 4; ++nt)
    #pragma unroll
    for (int r = 0; r < 8; ++r)
      w3v[nt * 8 + r] = W3[nt * 16 + 8 * half + r];

  char* h1row0 = smem + LDS_H1 + wave * H1_WAVEB + nl * H1_ROWB;
  char* h1row1 = h1row0 + H1_TILEB;

  const int tstride = gridDim.x * 16;     // 8 waves * 2 tiles per block
  for (int tile = (blockIdx.x * 8 + wave) * 2; tile < nTiles; tile += tstride) {
    int e0 = tile * 16 + nl;        if (e0 >= E) e0 = E - 1;
    int e1 = (tile + 1) * 16 + nl;  if (e1 >= E) e1 = E - 1;
    long long s0 = ei[e0], d0 = ei[E + e0];
    long long s1 = ei[e1], d1 = ei[E + e1];

    // ---- gather: B fragments (K = i + 16*half -> 32B contiguous runs) ----
    v16bf bf0[8], bf1[8];
    {
      const char* zs0 = (const char*)zb + s0 * 256;
      const char* zd0 = (const char*)zb + d0 * 256;
      const char* zs1 = (const char*)zb + s1 * 256;
      const char* zd1 = (const char*)zb + d1 * 256;
      #pragma unroll
      for (int kt = 0; kt < 8; ++kt) {
        int off = (kt & 3) * 64 + 32 * half;
        const char* p0 = ((kt < 4) ? zs0 : zd0) + off;
        const char* p1 = ((kt < 4) ? zs1 : zd1) + off;
        bf0[kt] = cat8(*(const v8bf*)p0, *(const v8bf*)(p0 + 16));
        bf1[kt] = cat8(*(const v8bf*)p1, *(const v8bf*)(p1 + 16));
      }
    }

    // ---- layer 1: two accumulator chains share each weight fragment ----
    #pragma unroll
    for (int nt = 0; nt < 16; ++nt) {
      v8f bias = *(const v8f*)(smem + LDS_B1 + nt * 64 + 32 * half);
      v8f acc0 = bias, acc1 = bias;
      #pragma unroll
      for (int kt = 0; kt < 8; ++kt) {
        const char* fb = smem + LDS_W1 + (nt * 8 + kt) * 1024;
        v16bf A = cat8(*(const v8bf*)(fb + lane * 16),
                       *(const v8bf*)(fb + 512 + lane * 16));
        acc0 = __builtin_amdgcn_wmma_f32_16x16x32_bf16(
            false, A, false, bf0[kt], (short)0, acc0, false, false);
        acc1 = __builtin_amdgcn_wmma_f32_16x16x32_bf16(
            false, A, false, bf1[kt], (short)0, acc1, false, false);
      }
      v8bf o0, o1;
      #pragma unroll
      for (int r = 0; r < 8; ++r) {        // D: M = nt*16+8*half+r, N = nl
        o0[r] = bf(fmaxf(acc0[r], 0.2f * acc0[r]));
        o1[r] = bf(fmaxf(acc1[r], 0.2f * acc1[r]));
      }
      *(v8bf*)(h1row0 + nt * 32 + 16 * half) = o0;
      *(v8bf*)(h1row1 + nt * 32 + 16 * half) = o1;
    }

    // ---- reload h1 as B fragments ----
    v16bf bh0[8], bh1[8];
    #pragma unroll
    for (int kt = 0; kt < 8; ++kt) {
      int off = kt * 64 + 32 * half;
      const char* p0 = h1row0 + off;
      const char* p1 = h1row1 + off;
      bh0[kt] = cat8(*(const v8bf*)p0, *(const v8bf*)(p0 + 16));
      bh1[kt] = cat8(*(const v8bf*)p1, *(const v8bf*)(p1 + 16));
    }

    // ---- layer 2 + in-register layer 3 partial dot ----
    float part0 = 0.f, part1 = 0.f;
    #pragma unroll
    for (int nt = 0; nt < 4; ++nt) {
      v8f bias = *(const v8f*)(smem + LDS_B2 + nt * 64 + 32 * half);
      v8f acc0 = bias, acc1 = bias;
      #pragma unroll
      for (int kt = 0; kt < 8; ++kt) {
        const char* fb = smem + LDS_W2 + (nt * 8 + kt) * 1024;
        v16bf A = cat8(*(const v8bf*)(fb + lane * 16),
                       *(const v8bf*)(fb + 512 + lane * 16));
        acc0 = __builtin_amdgcn_wmma_f32_16x16x32_bf16(
            false, A, false, bh0[kt], (short)0, acc0, false, false);
        acc1 = __builtin_amdgcn_wmma_f32_16x16x32_bf16(
            false, A, false, bh1[kt], (short)0, acc1, false, false);
      }
      #pragma unroll
      for (int r = 0; r < 8; ++r) {
        part0 += fmaxf(acc0[r], 0.2f * acc0[r]) * w3v[nt * 8 + r];
        part1 += fmaxf(acc1[r], 0.2f * acc1[r]) * w3v[nt * 8 + r];
      }
    }

    // combine the two half-lane partials (features 0..7 / 8..15 per tile)
    part0 += __shfl_xor(part0, 16);
    part1 += __shfl_xor(part1, 16);

    if (lane < 16) {
      int o0 = tile * 16 + lane;
      if (o0 < E) out[o0] = part0 + b3v;
      int o1 = (tile + 1) * 16 + lane;
      if (o1 < E) out[o1] = part1 + b3v;
    }
  }
}

extern "C" void kernel_launch(void* const* d_in, const int* in_sizes, int n_in,
                              void* d_out, int out_size, void* d_ws, size_t ws_size,
                              hipStream_t stream) {
  const float*     z  = (const float*)d_in[0];
  const long long* ei = (const long long*)d_in[1];
  const float*     W1 = (const float*)d_in[2];
  const float*     b1 = (const float*)d_in[3];
  const float*     W2 = (const float*)d_in[4];
  const float*     b2 = (const float*)d_in[5];
  const float*     W3 = (const float*)d_in[6];
  const float*     b3 = (const float*)d_in[7];
  float* out = (float*)d_out;

  int zn = in_sizes[0];                  // 100000 * 128
  int E  = in_sizes[1] / 2;              // edge_index is [2, E]
  int nTiles = (E + 15) / 16;            // E = 1e6 -> exactly 62500 tiles

  __bf16* wsbf = (__bf16*)d_ws;                          // weight fragments
  __bf16* zb   = (__bf16*)((char*)d_ws + WS_ZB_OFF);     // bf16 z copy

  prep_weights<<<320, 256, 0, stream>>>(W1, W2, wsbf);
  prep_z<<<(zn / 8 + 255) / 256, 256, 0, stream>>>(z, zb, zn);

  hipFuncSetAttribute((const void*)edge_mlp_kernel,
                      hipFuncAttributeMaxDynamicSharedMemorySize, LDS_TOTAL);

  int grid = 1024;                        // 16 tiles per block per iteration
  int maxg = (nTiles + 15) / 16;
  if (grid > maxg) grid = maxg;
  if (grid < 1) grid = 1;
  edge_mlp_kernel<<<grid, 256, LDS_TOTAL, stream>>>(zb, ei, b1, b2, W3, b3,
                                                    wsbf, out, E, nTiles);
}
